// AttentionModule_50087908606516
// MI455X (gfx1250) — matrix operand order repaired and checked by reading, not compiled, not run on hardware
//
#include <hip/hip_runtime.h>
#include <stdint.h>

// B=16, N=2048, D=512, O=512
#define NB   16
#define NSEQ 2048
#define DIM  512
#define BN   (NB * NSEQ)   // 32768

#define VST  72   // padded element stride of transposed V tile in LDS
#define WST  40   // padded element stride of transposed weight chunk in LDS

typedef __attribute__((ext_vector_type(16))) __bf16    v16bf;
typedef __attribute__((ext_vector_type(8)))  float     v8f;
typedef __attribute__((ext_vector_type(8)))  uint32_t  v8u;
typedef __attribute__((ext_vector_type(4)))  int       v4i;

// ---- scheduler pipelining directives (DS reads ahead of WMMAs) ------------
#if __has_builtin(__builtin_amdgcn_sched_group_barrier)
#define SCHED_DS_READ(n) __builtin_amdgcn_sched_group_barrier(0x100, (n), 0)
#define SCHED_WMMA(n)    __builtin_amdgcn_sched_group_barrier(0x008, (n), 0)
#else
#define SCHED_DS_READ(n)
#define SCHED_WMMA(n)
#endif
// pipeline pattern for a burst of 18 DS reads (A + 8xB) feeding 8 WMMAs
#define SCHED_BURST_8()                                                       \
  do {                                                                        \
    SCHED_DS_READ(6);                                                         \
    SCHED_WMMA(1); SCHED_DS_READ(2);                                          \
    SCHED_WMMA(1); SCHED_DS_READ(2);                                          \
    SCHED_WMMA(1); SCHED_DS_READ(2);                                          \
    SCHED_WMMA(1); SCHED_DS_READ(2);                                          \
    SCHED_WMMA(1); SCHED_DS_READ(2);                                          \
    SCHED_WMMA(1); SCHED_DS_READ(2);                                          \
    SCHED_WMMA(2);                                                            \
  } while (0)

// ---- float -> bf16 (packed) -----------------------------------------------
#if __has_builtin(__builtin_amdgcn_cvt_pk_bf16_f32)
static __device__ __forceinline__ uint32_t packbf(float a, float b) {
  auto p = __builtin_amdgcn_cvt_pk_bf16_f32(a, b);   // v_cvt_pk_bf16_f32
  return __builtin_bit_cast(uint32_t, p);
}
#else
// round-half-up + byte-permute pack: 2x v_add + 1x v_perm for two elements
static __device__ __forceinline__ uint32_t packbf(float a, float b) {
  uint32_t ua = __builtin_bit_cast(uint32_t, a) + 0x8000u;
  uint32_t ub = __builtin_bit_cast(uint32_t, b) + 0x8000u;
  return __builtin_amdgcn_perm(ub, ua, 0x07060302u);  // [ua.hi16, ub.hi16]
}
#endif
static __device__ __forceinline__ __bf16 f2bf1(float f) {
  return __builtin_bit_cast(__bf16, (uint16_t)packbf(f, f));
}
// Build a v16bf fragment from two 16-byte contiguous LDS runs (2x ds_load_b128).
static __device__ __forceinline__ v16bf frag2(const __bf16* lo, const __bf16* hi) {
  uint4 a = *(const uint4*)lo;
  uint4 b = *(const uint4*)hi;
  v8u t;
  t[0] = a.x; t[1] = a.y; t[2] = a.z; t[3] = a.w;
  t[4] = b.x; t[5] = b.y; t[6] = b.z; t[7] = b.w;
  return __builtin_bit_cast(v16bf, t);
}

// ---- async global->LDS staging (CDNA5 ASYNCcnt path), guarded -------------
#if __has_builtin(__builtin_amdgcn_global_load_async_to_lds_b128) && \
    __has_builtin(__builtin_amdgcn_s_wait_asynccnt)
#define HAVE_ASYNC_LDS 1
typedef __attribute__((address_space(1))) v4i* gv4_t;   // global int4*
typedef __attribute__((address_space(3))) v4i* lv4_t;   // LDS    int4*
// copy n16 * 16 bytes from global to LDS, cooperatively, async
static __device__ __forceinline__ void async_copy_b128(const void* g, void* l,
                                                       int n16, int tid, int nthr) {
  const char* gp = (const char*)g;
  char* lp = (char*)l;
  for (int i = tid; i < n16; i += nthr)
    __builtin_amdgcn_global_load_async_to_lds_b128(
        (gv4_t)(gp + (size_t)i * 16), (lv4_t)(lp + (size_t)i * 16), 0, 0);
  __builtin_amdgcn_s_wait_asynccnt(0);
}
#else
static __device__ __forceinline__ void async_copy_b128(const void* g, void* l,
                                                       int n16, int tid, int nthr) {
  const uint4* gp = (const uint4*)g;
  uint4* lp = (uint4*)l;
  for (int i = tid; i < n16; i += nthr) lp[i] = gp[i];
}
#endif

// ---------------------------------------------------------------------------
// Kernel 1: fused QKV projection.  out[wsel] = x @ kernel[wsel], stored bf16.
// ---------------------------------------------------------------------------
__global__ void __launch_bounds__(256, 1)
qkv_proj_kernel(const float* __restrict__ x,     // [BN][512] f32
                const float* __restrict__ w,     // [3][512][512] f32
                __bf16* __restrict__ qkv)        // [3][BN][512] bf16
{
  const int tid  = threadIdx.x;
  const int lane = tid & 31;
  const int wv   = tid >> 5;
  const int m0   = blockIdx.x * 32;    // row tile base in BN
  const int wsel = blockIdx.y;         // 0=q 1=k 2=v

  extern __shared__ char smem[];
  __bf16*   xs     = (__bf16*)smem;                      // [32][512]   32 KB
  __bf16*   wsmt   = (__bf16*)(smem + 32 * 512 * 2);     // [512][WST]  40 KB
  uint32_t* xs32   = (uint32_t*)xs;
  uint32_t* wsmt32 = (uint32_t*)wsmt;

  // stage x tile: float4 global loads -> packed bf16 dword LDS stores
  for (int i = tid; i < 32 * 128; i += 256) {
    int r = i >> 7, dg = i & 127;                        // row, group of 4 d
    float4 f = *(const float4*)(x + (size_t)(m0 + r) * DIM + dg * 4);
    int base = (r * 512 + dg * 4) >> 1;                  // dword index
    xs32[base]     = packbf(f.x, f.y);
    xs32[base + 1] = packbf(f.z, f.w);
  }

  const int h     = wv & 1;        // row half: rows h*16 .. h*16+15
  const int c     = wv >> 1;       // 128-col chunk (0..3)
  const int mrow  = lane & 15;
  const int khalf = lane >> 4;
  const int kbase = khalf * 16;

  v8f acc[8];
  for (int t = 0; t < 8; ++t)
    for (int e = 0; e < 8; ++e) acc[t][e] = 0.f;

  const __bf16* arow = xs + (size_t)(h * 16 + mrow) * 512;

  for (int kk = 0; kk < DIM; kk += 32) {
    __syncthreads();
    // stage weight K-chunk TRANSPOSED: wsmt[o][k], k = 0..31, packed pair dwords
    for (int i = tid; i < 16 * 128; i += 256) {
      int kp = i >> 7;        // k-pair 0..15
      int og = i & 127;       // group of 4 o
      const float* wp = w + (size_t)wsel * DIM * DIM + (size_t)(kk + 2 * kp) * DIM + og * 4;
      float4 fa = *(const float4*)wp;
      float4 fb = *(const float4*)(wp + DIM);
      const float fav[4] = {fa.x, fa.y, fa.z, fa.w};
      const float fbv[4] = {fb.x, fb.y, fb.z, fb.w};
#pragma unroll
      for (int j = 0; j < 4; ++j)                        // dword idx = o*(WST/2)+kp
        wsmt32[(og * 4 + j) * (WST / 2) + kp] = packbf(fav[j], fbv[j]);
    }
    __syncthreads();

    // A fragment + preload of all 8 B fragments, then 8-WMMA burst
    v16bf a = frag2(arow + kk + kbase, arow + kk + kbase + 8);
    v16bf bmv[8];
#pragma unroll
    for (int t = 0; t < 8; ++t) {
      const __bf16* bcol = wsmt + (size_t)(c * 128 + t * 16 + mrow) * WST;
      bmv[t] = frag2(bcol + kbase, bcol + kbase + 8);    // K = kbase+0..15 contiguous
    }
#pragma unroll
    for (int t = 0; t < 8; ++t)
      acc[t] = __builtin_amdgcn_wmma_f32_16x16x32_bf16(false, a, false, bmv[t],
                                                       (short)0, acc[t], false, false);
    SCHED_BURST_8();   // pipeline DS reads one stage ahead of WMMAs
  }

  // epilogue: scatter bf16 result into LDS, then coalesced b128 copy-out
  __syncthreads();
  __bf16* obuf = (__bf16*)smem;    // reuse xs region, [32][512] bf16
#pragma unroll
  for (int t = 0; t < 8; ++t) {
#pragma unroll
    for (int r = 0; r < 8; ++r) {
      int m = h * 16 + r + 8 * khalf;
      int n = c * 128 + t * 16 + mrow;
      obuf[m * 512 + n] = f2bf1(acc[t][r]);
    }
  }
  __syncthreads();
  {
    const uint4* ob4 = (const uint4*)obuf;
    uint4* og4 = (uint4*)(qkv + (size_t)wsel * BN * DIM + (size_t)m0 * DIM);
    for (int i = tid; i < 32 * 512 / 8; i += 256) og4[i] = ob4[i];
  }
}

// ---------------------------------------------------------------------------
// Kernel 2: flash attention over bf16 q/k/v with f32 accumulation.
// ---------------------------------------------------------------------------
__global__ void __launch_bounds__(256, 1)
flash_attn_kernel(const __bf16* __restrict__ qb,
                  const __bf16* __restrict__ kb,
                  const __bf16* __restrict__ vb,
                  float* __restrict__ out)
{
  const int tid  = threadIdx.x;
  const int lane = tid & 31;
  const int wv   = tid >> 5;
  const int bidx = blockIdx.y;
  const int q0   = blockIdx.x * 32;

  extern __shared__ char smem[];
  __bf16* qs   = (__bf16*)smem;                                  // 32 KB @ 0
  __bf16* ks   = (__bf16*)(smem + 32768);                        // 64 KB
  __bf16* vst  = (__bf16*)(smem + 98304);                        // [512][VST] 72 KB
  float*  ss   = (float*)(smem + 172032);                        // [32][64] 8 KB
  float*  rowm = (float*)(smem + 180224);
  float*  rowl = (float*)(smem + 180352);
  float*  rowa = (float*)(smem + 180480);
  __bf16* pb   = (__bf16*)(smem + 180608);                       // [32][64] 4 KB
  float*  red  = (float*)(smem + 184704);                        // [32][8] 1 KB
  uint32_t* vst32 = (uint32_t*)vst;
  uint32_t* pb32  = (uint32_t*)pb;

  // stage q tile (bf16) via async b128 global->LDS copies
  async_copy_b128(qb + (size_t)(bidx * NSEQ + q0) * DIM, qs, 32 * 512 / 8, tid, 256);
  if (tid < 32) { rowm[tid] = -1e30f; rowl[tid] = 0.f; }
  __syncthreads();

  const int h     = wv & 1;    // query half
  const int c     = wv >> 1;   // scores: key subtile 0..3 | output: 128-col chunk 0..3
  const int mrow  = lane & 15;
  const int khalf = lane >> 4;
  const int kbase = khalf * 16;
  const float scale = 0.04419417382415922f;  // 1/sqrt(512)

  // cache all 16 Q fragments in registers for the whole kernel
  v16bf qfrag[16];
  {
    const __bf16* qrow = qs + (size_t)(h * 16 + mrow) * 512;
#pragma unroll
    for (int i = 0; i < 16; ++i)
      qfrag[i] = frag2(qrow + i * 32 + kbase, qrow + i * 32 + kbase + 8);
  }

  v8f acc[8];
  for (int t = 0; t < 8; ++t)
    for (int e = 0; e < 8; ++e) acc[t][e] = 0.f;

  for (int kt = 0; kt < NSEQ; kt += 64) {
    __syncthreads();
    // K tile row-major (async b128 copies)
    async_copy_b128(kb + (size_t)(bidx * NSEQ + kt) * DIM, ks, 64 * 512 / 8, tid, 256);
    {
      // V tile TRANSPOSED: vst[o][key]; key pairs packed with v_perm_b32
      const uint4* v4 = (const uint4*)(vb + (size_t)(bidx * NSEQ + kt) * DIM);
      for (int i = tid; i < 32 * 64; i += 256) {
        int kp = i >> 6;              // key pair 0..31
        int og = i & 63;              // group of 8 o
        uint4 ua = v4[(2 * kp) * 64 + og];
        uint4 ub = v4[(2 * kp + 1) * 64 + og];
        const uint32_t aw[4] = {ua.x, ua.y, ua.z, ua.w};
        const uint32_t bw[4] = {ub.x, ub.y, ub.z, ub.w};
        int o0 = og * 8;
#pragma unroll
        for (int j = 0; j < 4; ++j) {  // dword idx = o*(VST/2)+kp
          uint32_t d0 = __builtin_amdgcn_perm(bw[j], aw[j], 0x05040100u); // lo halves
          uint32_t d1 = __builtin_amdgcn_perm(bw[j], aw[j], 0x07060302u); // hi halves
          vst32[(o0 + 2 * j)     * (VST / 2) + kp] = d0;
          vst32[(o0 + 2 * j + 1) * (VST / 2) + kp] = d1;
        }
      }
    }
    __syncthreads();

    if (kt + 64 < NSEQ) {  // warm caches for next tile (global_prefetch_b8)
      __builtin_prefetch(kb + (size_t)(bidx * NSEQ + kt + 64) * DIM + tid * 64, 0, 1);
      __builtin_prefetch(vb + (size_t)(bidx * NSEQ + kt + 64) * DIM + tid * 64, 0, 1);
    }

    // ---- scores: each wave computes a 16x16 tile of S = Q K^T (K-dim = 512)
    v8f sacc;
    for (int e = 0; e < 8; ++e) sacc[e] = 0.f;
    const __bf16* krow = ks + (size_t)(c * 16 + mrow) * 512;  // lane = key col n
#pragma unroll 4
    for (int kk = 0; kk < 16; ++kk) {
      v16bf bm = frag2(krow + kk * 32 + kbase, krow + kk * 32 + kbase + 8);
      sacc = __builtin_amdgcn_wmma_f32_16x16x32_bf16(false, qfrag[kk], false, bm,
                                                     (short)0, sacc, false, false);
      // per unrolled 4-iter body: pipeline 8 DS reads around 4 WMMAs
      if ((kk & 3) == 3) {
        SCHED_DS_READ(4);
        SCHED_WMMA(1); SCHED_DS_READ(2);
        SCHED_WMMA(1); SCHED_DS_READ(2);
        SCHED_WMMA(2);
      }
    }
#pragma unroll
    for (int r = 0; r < 8; ++r)
      ss[(h * 16 + r + 8 * khalf) * 64 + c * 16 + mrow] = sacc[r] * scale;
    __syncthreads();

    // ---- online softmax, parallel across all 256 threads (8 per row) ----
    {
      int row = tid >> 3, seg = tid & 7;
      const float* srow = ss + row * 64 + seg * 8;
      float m8 = srow[0];
#pragma unroll
      for (int j = 1; j < 8; ++j) m8 = fmaxf(m8, srow[j]);
      red[tid] = m8;
    }
    __syncthreads();
    if (tid < 32) {                  // per-row: new max + alpha
      float mo = rowm[tid];
      float tm = mo;
#pragma unroll
      for (int j = 0; j < 8; ++j) tm = fmaxf(tm, red[tid * 8 + j]);
      rowm[tid] = tm;
      rowa[tid] = __expf(mo - tm);
    }
    __syncthreads();
    {
      int row = tid >> 3, seg = tid & 7;
      float tm = rowm[row];
      const float* srow = ss + row * 64 + seg * 8;
      uint32_t* pr = pb32 + row * 32 + seg * 4;
      float s8 = 0.f;
#pragma unroll
      for (int j = 0; j < 4; ++j) {
        float p0 = __expf(srow[2 * j]     - tm);
        float p1 = __expf(srow[2 * j + 1] - tm);
        s8 += p0 + p1;
        pr[j] = packbf(p0, p1);
      }
      red[tid] = s8;
    }
    __syncthreads();
    if (tid < 32) {                  // per-row: running sum
      float l = rowl[tid] * rowa[tid];
#pragma unroll
      for (int j = 0; j < 8; ++j) l += red[tid * 8 + j];
      rowl[tid] = l;
    }
    __syncthreads();

    // ---- rescale accumulators, then O += P(16x64,bf16) @ V(64x128-chunk)
    float al[8];
#pragma unroll
    for (int r = 0; r < 8; ++r) al[r] = rowa[h * 16 + r + 8 * khalf];
#pragma unroll
    for (int t = 0; t < 8; ++t)
#pragma unroll
      for (int r = 0; r < 8; ++r) acc[t][r] *= al[r];

    const __bf16* prow = pb + (size_t)(h * 16 + mrow) * 64;
#pragma unroll
    for (int k2 = 0; k2 < 2; ++k2) {   // two K=32 slices of the 64 keys
      v16bf a = frag2(prow + k2 * 32 + kbase, prow + k2 * 32 + kbase + 8);
      v16bf bmv[8];
#pragma unroll
      for (int t = 0; t < 8; ++t) {
        const __bf16* vcol = vst + (size_t)(c * 128 + t * 16 + mrow) * VST + k2 * 32;
        bmv[t] = frag2(vcol + kbase, vcol + kbase + 8);  // keys contiguous (K-major)
      }
#pragma unroll
      for (int t = 0; t < 8; ++t)
        acc[t] = __builtin_amdgcn_wmma_f32_16x16x32_bf16(false, a, false, bmv[t],
                                                         (short)0, acc[t], false, false);
      SCHED_BURST_8();   // pipeline DS reads one stage ahead of WMMAs
    }
  }

  // epilogue: scatter f32 result into LDS (reuse ks region), b128 copy-out
  __syncthreads();
  float linv[8];
#pragma unroll
  for (int r = 0; r < 8; ++r) linv[r] = 1.0f / rowl[h * 16 + r + 8 * khalf];
  float* obuf = (float*)(smem + 32768);   // [32][512] f32 = 64 KB (= ks region)
#pragma unroll
  for (int t = 0; t < 8; ++t) {
#pragma unroll
    for (int r = 0; r < 8; ++r) {
      int m = h * 16 + r + 8 * khalf;
      int n = c * 128 + t * 16 + mrow;
      obuf[m * 512 + n] = acc[t][r] * linv[r];
    }
  }
  __syncthreads();
  {
    const uint4* ob4 = (const uint4*)obuf;
    uint4* og4 = (uint4*)(out + ((size_t)bidx * NSEQ + q0) * DIM);
    for (int i = tid; i < 32 * 512 / 4; i += 256) og4[i] = ob4[i];
  }
}

// ---------------------------------------------------------------------------
extern "C" void kernel_launch(void* const* d_in, const int* in_sizes, int n_in,
                              void* d_out, int out_size, void* d_ws, size_t ws_size,
                              hipStream_t stream) {
  (void)in_sizes; (void)n_in; (void)out_size; (void)ws_size;
  const float* x = (const float*)d_in[0];   // [16][2048][512] f32
  const float* w = (const float*)d_in[1];   // [3][512][512]   f32
  float* out = (float*)d_out;               // [16][2048][512] f32

  // workspace: q,k,v as bf16, contiguous: 3 * 32768 * 512 * 2 B = 100.7 MB
  __bf16* qkv = (__bf16*)d_ws;

  const size_t smem_proj = 32 * 512 * 2 + 512 * WST * 2;            // 73728 B
  const size_t smem_attn = 184704 + 1024;                           // 185728 B

  (void)hipFuncSetAttribute((const void*)qkv_proj_kernel,
                            hipFuncAttributeMaxDynamicSharedMemorySize, (int)smem_proj);
  (void)hipFuncSetAttribute((const void*)flash_attn_kernel,
                            hipFuncAttributeMaxDynamicSharedMemorySize, (int)smem_attn);

  dim3 gp(BN / 32, 3);
  qkv_proj_kernel<<<gp, dim3(256), smem_proj, stream>>>(x, w, qkv);

  dim3 ga(NSEQ / 32, NB);
  flash_attn_kernel<<<ga, dim3(256), smem_attn, stream>>>(
      qkv,                                  // q
      qkv + (size_t)BN * DIM,               // k
      qkv + (size_t)2 * BN * DIM,           // v
      out);
}
